// TransformerBlock_49821620633660
// MI455X (gfx1250) — matrix operand order repaired and checked by reading, not compiled, not run on hardware
//
#include <hip/hip_runtime.h>
#include <hip/hip_bf16.h>

// ---------------------------------------------------------------------------
// Transformer block for gfx1250 (MI455X): bf16 WMMA everywhere, flash attn,
// double-buffered async global->LDS tile staging (ASYNCcnt path) in the GEMMs.
// ---------------------------------------------------------------------------

typedef __attribute__((ext_vector_type(16))) __bf16 v16bf;
typedef __attribute__((ext_vector_type(8)))  __bf16 v8bf;
typedef __attribute__((ext_vector_type(8)))  float  v8f;

static constexpr int B_  = 4;
static constexpr int S_  = 2048;
static constexpr int D_  = 1024;
static constexpr int H_  = 16;
static constexpr int HD_ = 64;
static constexpr int M_  = B_ * S_;     // 8192 rows
static constexpr int D3_ = 3 * D_;      // 3072
static constexpr int D4_ = 4 * D_;      // 4096

// ---------------- fragment helpers (CDNA5 16-bit WMMA VGPR layouts) --------

__device__ __forceinline__ v16bf cat8(v8bf a, v8bf b) {
  return __builtin_shufflevector(a, b, 0,1,2,3,4,5,6,7,8,9,10,11,12,13,14,15);
}

// A-matrix 16x32 bf16 fragment: lane holds row M=lane&15;
// elements 0..15 = K (sel*16 .. sel*16+15), contiguous
__device__ __forceinline__ v16bf frag_a(const __bf16* rowp, int sel) {
  v8bf lo = *(const v8bf*)(rowp + sel * 16);
  v8bf hi = *(const v8bf*)(rowp + sel * 16 + 8);
  return cat8(lo, hi);
}

// B-matrix 32x16 bf16 fragment: lane holds col N=lane&15;
// elements 0..15 = K (sel*16 .. sel*16+15), contiguous
__device__ __forceinline__ v16bf frag_b(const __bf16* rowp, int sel) {
  const v8bf* p = (const v8bf*)(rowp + sel * 16);
  return cat8(p[0], p[1]);
}

__device__ __forceinline__ v8f wmma_bf16(v16bf a, v16bf b, v8f c) {
  return __builtin_amdgcn_wmma_f32_16x16x32_bf16(
      /*neg_a=*/false, a, /*neg_b=*/false, b,
      /*c_mod=*/(short)0, c, /*reuse_a=*/false, /*reuse_b=*/false);
}

// async global->LDS 16-byte copy (gfx1250, tracked by ASYNCcnt)
__device__ __forceinline__ void async_copy_b128(const __bf16* gptr, __bf16* lptr) {
  unsigned lds_off = (unsigned)(unsigned long long)lptr;   // addr[31:0] = LDS offset
  unsigned long long gaddr = (unsigned long long)gptr;
  asm volatile("global_load_async_to_lds_b128 %0, %1, off"
               :: "v"(lds_off), "v"(gaddr) : "memory");
}
__device__ __forceinline__ void wait_async0() {
  asm volatile("s_wait_asynccnt 0" ::: "memory");
}

__device__ __forceinline__ float gelu_tanh_f(float x) {
  const float c = 0.7978845608028654f;  // sqrt(2/pi)
  float u = c * (x + 0.044715f * x * x * x);
  return 0.5f * x * (1.0f + tanhf(u));
}

// half-wave (16-lane) reductions: C/D layout puts one row's N values in one half
__device__ __forceinline__ float hmax16(float v) {
  v = fmaxf(v, __shfl_xor(v, 1, 32));
  v = fmaxf(v, __shfl_xor(v, 2, 32));
  v = fmaxf(v, __shfl_xor(v, 4, 32));
  v = fmaxf(v, __shfl_xor(v, 8, 32));
  return v;
}
__device__ __forceinline__ float hsum16(float v) {
  v += __shfl_xor(v, 1, 32);
  v += __shfl_xor(v, 2, 32);
  v += __shfl_xor(v, 4, 32);
  v += __shfl_xor(v, 8, 32);
  return v;
}

// ---------------- LayerNorm (fp32 in -> bf16 out), one wave per row --------

__global__ __launch_bounds__(256)
void layernorm_bf16_kernel(const float* __restrict__ x,
                           const float* __restrict__ g,
                           const float* __restrict__ s,
                           __bf16* __restrict__ out) {
  const int wid  = threadIdx.x >> 5;
  const int lane = threadIdx.x & 31;
  const int row  = blockIdx.x * 8 + wid;
  const float* xr = x + (size_t)row * D_;

  float v[32];
  float sum = 0.f;
#pragma unroll
  for (int i = 0; i < 32; ++i) { v[i] = xr[lane + i * 32]; sum += v[i]; }
#pragma unroll
  for (int m = 1; m < 32; m <<= 1) sum += __shfl_xor(sum, m, 32);
  const float mean = sum * (1.0f / (float)D_);

  float vs = 0.f;
#pragma unroll
  for (int i = 0; i < 32; ++i) { float d = v[i] - mean; vs += d * d; }
#pragma unroll
  for (int m = 1; m < 32; m <<= 1) vs += __shfl_xor(vs, m, 32);
  const float inv = rsqrtf(vs * (1.0f / (float)D_) + 1e-5f);

  __bf16* orow = out + (size_t)row * D_;
#pragma unroll
  for (int i = 0; i < 32; ++i) {
    int c = lane + i * 32;
    orow[c] = (__bf16)(g[c] * ((v[i] - mean) * inv) + s[c]);
  }
}

// ---------------- weight convert + transpose: W[K][N] f32 -> Wt[N][K] bf16 -

__global__ __launch_bounds__(256)
void convert_wt_kernel(const float* __restrict__ W, __bf16* __restrict__ Wt,
                       int K, int N) {
  int t = blockIdx.x * 256 + threadIdx.x;  // < N*K (<= 4.2M)
  int n = t / K;
  int k = t - n * K;
  Wt[t] = (__bf16)W[(size_t)k * N + n];
}

// ---------------- V transpose: qkv bf16 -> Vt[b][h][d][S] bf16 -------------

__global__ __launch_bounds__(256)
void transpose_v_kernel(const __bf16* __restrict__ qkv,
                        __bf16* __restrict__ Vt) {
  int t = blockIdx.x * 256 + threadIdx.x;  // < B*H*64*S = 2^23
  int s = t & (S_ - 1);
  int d = (t >> 11) & 63;
  int h = (t >> 17) & 15;
  int b = t >> 21;
  Vt[t] = qkv[((size_t)(b * S_ + s)) * D3_ + 2 * D_ + h * HD_ + d];
}

// ---------------- tiled bf16 GEMM with fused epilogues ---------------------
// C[M,N] = A[M,K](bf16) x Wt[N,K](bf16), 128x128 block tile, BK=32,
// 8 waves -> 32x64 per wave (2x4 v_wmma_f32_16x16x32_bf16 per K-step).
// Double-buffered LDS tiles staged with global_load_async_to_lds_b128:
// the copy of tile t+1 overlaps the 8 WMMAs of tile t.

enum { EPI_BF16 = 0, EPI_BIAS_RESID_F32 = 1, EPI_BIAS_GELU_BF16 = 2 };

template <int EPI>
__global__ __launch_bounds__(256)
void gemm_bf16_kernel(const __bf16* __restrict__ A,
                      const __bf16* __restrict__ Wt,
                      const float* __restrict__ bias,
                      const float* __restrict__ resid,
                      float* __restrict__ outF,
                      __bf16* __restrict__ outB,
                      int N, int K) {
  __shared__ __align__(16) __bf16 As[2][128 * 32];
  __shared__ __align__(16) __bf16 Bs[2][128 * 32];

  const int tid  = threadIdx.x;
  const int lane = tid & 31;
  const int wid  = tid >> 5;
  const int sel  = lane >> 4;
  const int lrow = lane & 15;
  const int wm   = (wid & 3) * 32;     // 4 wave-rows
  const int wn   = (wid >> 2) * 64;    // 2 wave-cols
  const int m0   = blockIdx.y * 128;
  const int n0   = blockIdx.x * 128;

  v8f acc[2][4];
#pragma unroll
  for (int i = 0; i < 2; ++i)
#pragma unroll
    for (int j = 0; j < 4; ++j) acc[i][j] = (v8f){0,0,0,0,0,0,0,0};

  // staging map: each thread copies one 32B row-half of As and of Bs
  const int srow = tid >> 1;            // 0..127
  const int sk   = (tid & 1) * 16;      // 0 or 16 elements

  auto stage = [&](int buf, int k0) {
    const __bf16* ag = A  + (size_t)(m0 + srow) * K + k0 + sk;
    const __bf16* bg = Wt + (size_t)(n0 + srow) * K + k0 + sk;
    async_copy_b128(ag,     As[buf] + srow * 32 + sk);
    async_copy_b128(ag + 8, As[buf] + srow * 32 + sk + 8);
    async_copy_b128(bg,     Bs[buf] + srow * 32 + sk);
    async_copy_b128(bg + 8, Bs[buf] + srow * 32 + sk + 8);
  };

  stage(0, 0);
  wait_async0();
  __syncthreads();

  int buf = 0;
  for (int k0 = 0; k0 < K; k0 += 32) {
    if (k0 + 32 < K) stage(buf ^ 1, k0 + 32);   // overlap copy with compute

    v16bf af[2], bfv[4];
#pragma unroll
    for (int i = 0; i < 2; ++i)
      af[i] = frag_a(As[buf] + (wm + i * 16 + lrow) * 32, sel);
#pragma unroll
    for (int j = 0; j < 4; ++j)
      bfv[j] = frag_b(Bs[buf] + (wn + j * 16 + lrow) * 32, sel);

#pragma unroll
    for (int i = 0; i < 2; ++i)
#pragma unroll
      for (int j = 0; j < 4; ++j)
        acc[i][j] = wmma_bf16(af[i], bfv[j], acc[i][j]);

    wait_async0();        // next tile fully in LDS
    __syncthreads();      // all waves done reading buf / writing buf^1
    buf ^= 1;
  }

#pragma unroll
  for (int i = 0; i < 2; ++i) {
#pragma unroll
    for (int j = 0; j < 4; ++j) {
#pragma unroll
      for (int r = 0; r < 8; ++r) {
        const int row = m0 + wm + i * 16 + r + sel * 8;
        const int col = n0 + wn + j * 16 + lrow;
        float v = acc[i][j][r];
        if (EPI == EPI_BIAS_RESID_F32) {
          v += bias[col] + resid[(size_t)row * N + col];
          outF[(size_t)row * N + col] = v;
        } else if (EPI == EPI_BIAS_GELU_BF16) {
          v = gelu_tanh_f(v + bias[col]);
          outB[(size_t)row * N + col] = (__bf16)v;
        } else {
          outB[(size_t)row * N + col] = (__bf16)v;
        }
      }
    }
  }
}

// ---------------- flash attention: one wave per (b, h, 16 q-rows) ----------

__global__ __launch_bounds__(128)
void attn_kernel(const __bf16* __restrict__ qkv,
                 const __bf16* __restrict__ Vt,
                 __bf16* __restrict__ ctx) {
  __shared__ __align__(16) __bf16 Pt[4][16 * 32];   // per-wave P tile

  const int tid  = threadIdx.x;
  const int lane = tid & 31;
  const int wid  = tid >> 5;
  const int sel  = lane >> 4;
  const int lrow = lane & 15;
  __bf16* P = Pt[wid];

  const int w  = blockIdx.x * 4 + wid;   // 0..8191
  const int qt = w & 127;
  const int h  = (w >> 7) & 15;
  const int b  = w >> 11;
  const int qbase = qt * 16;

  // Q fragments (A layout), two K=32 chunks over HD=64
  const __bf16* qp = qkv + ((size_t)(b * S_ + qbase + lrow)) * D3_ + h * HD_;
  v16bf qf0 = cat8(*(const v8bf*)(qp + sel * 16),
                   *(const v8bf*)(qp + sel * 16 + 8));
  v16bf qf1 = cat8(*(const v8bf*)(qp + 32 + sel * 16),
                   *(const v8bf*)(qp + 40 + sel * 16));

  v8f o[4];
#pragma unroll
  for (int c = 0; c < 4; ++c) o[c] = (v8f){0,0,0,0,0,0,0,0};
  float mrun[8], lrun[8];
#pragma unroll
  for (int r = 0; r < 8; ++r) { mrun[r] = -1e30f; lrun[r] = 0.f; }

  const __bf16* vbase = Vt + ((size_t)((b * H_ + h) * HD_)) * S_;

  for (int kb = 0; kb < qbase + 16; kb += 32) {
    // ---- scores: 16 q-rows x 32 keys, K=HD=64 (2 chunks x 2 key tiles)
    const __bf16* k0p = qkv + ((size_t)(b * S_ + kb + lrow)) * D3_ + D_ + h * HD_;
    const __bf16* k1p = k0p + (size_t)16 * D3_;
    v8f s0 = (v8f){0,0,0,0,0,0,0,0};
    v8f s1 = (v8f){0,0,0,0,0,0,0,0};
#pragma unroll
    for (int kc = 0; kc < 2; ++kc) {
      v16bf bk0 = cat8(*(const v8bf*)(k0p + kc * 32 + sel * 16),
                       *(const v8bf*)(k0p + kc * 32 + sel * 16 + 8));
      v16bf bk1 = cat8(*(const v8bf*)(k1p + kc * 32 + sel * 16),
                       *(const v8bf*)(k1p + kc * 32 + sel * 16 + 8));
      v16bf a = kc ? qf1 : qf0;
      s0 = wmma_bf16(a, bk0, s0);
      s1 = wmma_bf16(a, bk1, s1);
    }

    // ---- online softmax (scale 1/sqrt(64)=0.125, causal mask)
#pragma unroll
    for (int r = 0; r < 8; ++r) {
      const int qrow = qbase + r + sel * 8;
      float a = s0[r] * 0.125f;
      float c = s1[r] * 0.125f;
      if (kb + lrow      > qrow) a = -1e30f;
      if (kb + 16 + lrow > qrow) c = -1e30f;
      float bm   = hmax16(fmaxf(a, c));
      float mnew = fmaxf(mrun[r], bm);
      float scl  = __expf(mrun[r] - mnew);
      float e0   = __expf(a - mnew);
      float e1   = __expf(c - mnew);
      float rs   = hsum16(e0 + e1);
      lrun[r] = lrun[r] * scl + rs;
      mrun[r] = mnew;
#pragma unroll
      for (int c2 = 0; c2 < 4; ++c2) o[c2][r] *= scl;
      P[(r + sel * 8) * 32 + lrow]      = (__bf16)e0;
      P[(r + sel * 8) * 32 + 16 + lrow] = (__bf16)e1;
    }
    asm volatile("s_wait_dscnt 0" ::: "memory");

    // ---- ctx += P(16x32) @ V(32x64): reload P in A layout, V^T is B layout
    v16bf pa = cat8(*(const v8bf*)(P + lrow * 32 + sel * 16),
                    *(const v8bf*)(P + lrow * 32 + sel * 16 + 8));
#pragma unroll
    for (int c2 = 0; c2 < 4; ++c2) {
      const __bf16* vp = vbase + (size_t)(c2 * 16 + lrow) * S_ + kb + sel * 16;
      v16bf bv = cat8(*(const v8bf*)vp, *(const v8bf*)(vp + 8));
      o[c2] = wmma_bf16(pa, bv, o[c2]);
    }
  }

  // ---- normalize and store ctx (bf16) for the Wout GEMM
#pragma unroll
  for (int c2 = 0; c2 < 4; ++c2) {
#pragma unroll
    for (int r = 0; r < 8; ++r) {
      const int row = qbase + r + sel * 8;
      ctx[((size_t)(b * S_ + row)) * D_ + h * HD_ + c2 * 16 + lrow] =
          (__bf16)(o[c2][r] / lrun[r]);
    }
  }
}

// ---------------------------------------------------------------------------

extern "C" void kernel_launch(void* const* d_in, const int* in_sizes, int n_in,
                              void* d_out, int out_size, void* d_ws,
                              size_t ws_size, hipStream_t stream) {
  const float* x    = (const float*)d_in[0];
  const float* Wqkv = (const float*)d_in[1];
  const float* Wout = (const float*)d_in[2];
  const float* bout = (const float*)d_in[3];
  const float* W1   = (const float*)d_in[4];
  const float* b1   = (const float*)d_in[5];
  const float* W2   = (const float*)d_in[6];
  const float* b2   = (const float*)d_in[7];
  const float* g1   = (const float*)d_in[8];
  const float* s1   = (const float*)d_in[9];
  const float* g2   = (const float*)d_in[10];
  const float* s2   = (const float*)d_in[11];
  float* out = (float*)d_out;

  char* ws = (char*)d_ws;
  size_t off = 0;
  auto alloc = [&](size_t bytes) -> char* {
    char* p = ws + off;
    off += (bytes + 255) & ~(size_t)255;
    return p;
  };
  __bf16* h1     = (__bf16*)alloc((size_t)M_ * D_ * 2);    // ln1 out
  __bf16* Wqkv_t = (__bf16*)alloc((size_t)D3_ * D_ * 2);
  __bf16* qkvb   = (__bf16*)alloc((size_t)M_ * D3_ * 2);
  __bf16* Vtb    = (__bf16*)alloc((size_t)B_ * H_ * HD_ * S_ * 2);
  __bf16* ctxb   = (__bf16*)alloc((size_t)M_ * D_ * 2);
  __bf16* Wout_t = (__bf16*)alloc((size_t)D_ * D_ * 2);
  __bf16* h2     = (__bf16*)alloc((size_t)M_ * D_ * 2);    // ln2 out
  __bf16* W1_t   = (__bf16*)alloc((size_t)D4_ * D_ * 2);
  __bf16* midb   = (__bf16*)alloc((size_t)M_ * D4_ * 2);   // gelu(h2@W1+b1)
  __bf16* W2_t   = (__bf16*)alloc((size_t)D_ * D4_ * 2);
  (void)ws_size; (void)in_sizes; (void)n_in; (void)out_size;

  const dim3 blk256(256), blk128(128);

  // 1) weight convert+transpose to [N][K] bf16
  convert_wt_kernel<<<(D_ * D3_) / 256, blk256, 0, stream>>>(Wqkv, Wqkv_t, D_, D3_);
  convert_wt_kernel<<<(D_ * D_)  / 256, blk256, 0, stream>>>(Wout, Wout_t, D_, D_);
  convert_wt_kernel<<<(D_ * D4_) / 256, blk256, 0, stream>>>(W1,   W1_t,   D_, D4_);
  convert_wt_kernel<<<(D4_ * D_) / 256, blk256, 0, stream>>>(W2,   W2_t,   D4_, D_);

  // 2) ln1(x) -> h1 bf16
  layernorm_bf16_kernel<<<M_ / 8, blk256, 0, stream>>>(x, g1, s1, h1);

  // 3) qkv = h1 @ Wqkv  (bf16 out)
  gemm_bf16_kernel<EPI_BF16><<<dim3(D3_ / 128, M_ / 128), blk256, 0, stream>>>(
      h1, Wqkv_t, nullptr, nullptr, nullptr, qkvb, D3_, D_);

  // 4) V^T for the ctx GEMM B-fragments
  transpose_v_kernel<<<(B_ * H_ * HD_ * S_) / 256, blk256, 0, stream>>>(qkvb, Vtb);

  // 5) flash attention -> ctx bf16
  attn_kernel<<<(B_ * H_ * (S_ / 16)) / 4, blk128, 0, stream>>>(qkvb, Vtb, ctxb);

  // 6) x_attn = x + ctx @ Wout + bout  -> d_out (fp32)
  gemm_bf16_kernel<EPI_BIAS_RESID_F32><<<dim3(D_ / 128, M_ / 128), blk256, 0, stream>>>(
      ctxb, Wout_t, bout, x, out, nullptr, D_, D_);

  // 7) ln2(x_attn) -> h2 bf16
  layernorm_bf16_kernel<<<M_ / 8, blk256, 0, stream>>>(out, g2, s2, h2);

  // 8) mid = gelu(h2 @ W1 + b1) -> bf16
  gemm_bf16_kernel<EPI_BIAS_GELU_BF16><<<dim3(D4_ / 128, M_ / 128), blk256, 0, stream>>>(
      h2, W1_t, b1, nullptr, nullptr, midb, D4_, D_);

  // 9) out = x_attn + mid @ W2 + b2 -> d_out (fp32)
  gemm_bf16_kernel<EPI_BIAS_RESID_F32><<<dim3(D_ / 128, M_ / 128), blk256, 0, stream>>>(
      midb, W2_t, b2, out, out, nullptr, D_, D4_);
}
